// AttentionBlock_87643102642422
// MI455X (gfx1250) — matrix operand order repaired and checked by reading, not compiled, not run on hardware
//
#include <hip/hip_runtime.h>
#include <hip/hip_bf16.h>

typedef __attribute__((ext_vector_type(16))) _Float16 v16h;
typedef __attribute__((ext_vector_type(8)))  _Float16 v8h;
typedef __attribute__((ext_vector_type(8)))  float    v8f;

#define WMMA_F32_F16(a, b, c) \
  __builtin_amdgcn_wmma_f32_16x16x32_f16(false, (a), false, (b), (short)0, (c), false, false)

static __device__ __forceinline__ int lane_id() { return threadIdx.x & 31; }

// A fragment (16x32 f16, M x K), source row-major, row stride ld (halfs).
// lane<16: row=lane, halfs i = K=i
// lane>=16: row=lane-16, halfs i = K=16+i
static __device__ __forceinline__ v16h load_a_frag(const _Float16* p, int ld) {
  const int lane = lane_id();
  const int row  = lane & 15;
  const int kb   = (lane & 16) ? 16 : 0;
  const _Float16* q = p + (size_t)row * ld + kb;
  v8h lo = *(const v8h*)(q);
  v8h hi = *(const v8h*)(q + 8);
  v16h a;
#pragma unroll
  for (int i = 0; i < 8; ++i) { a[i] = lo[i]; a[8 + i] = hi[i]; }
  return a;
}

// B fragment (32x16, K x N) where the source holds B^T row-major [N x K], row stride ld.
// lane<16: col=lane, halfs i = K=i ; lane>=16: col=lane-16, halfs i = K=16+i.
// Contiguous 32B per lane.
static __device__ __forceinline__ v16h load_bT_frag(const _Float16* p, int ld) {
  const int lane = lane_id();
  const int col  = lane & 15;
  const int k0   = (lane & 16) ? 16 : 0;
  const _Float16* q = p + (size_t)col * ld + k0;
  v8h lo = *(const v8h*)(q);
  v8h hi = *(const v8h*)(q + 8);
  v16h b;
#pragma unroll
  for (int i = 0; i < 8; ++i) { b[i] = lo[i]; b[8 + i] = hi[i]; }
  return b;
}

__global__ void cvt_f32_to_f16(const float* __restrict__ in,
                               _Float16* __restrict__ out, int n) {
  int i = blockIdx.x * blockDim.x + threadIdx.x;
  if (i < n) out[i] = (_Float16)in[i];
}

// out[n*K + k] = (f16) in[k*N + n]  -- weight transpose+convert (one-shot, L2 resident)
__global__ void cvt_transpose_f16(const float* __restrict__ in,
                                  _Float16* __restrict__ out, int K, int N) {
  int i = blockIdx.x * blockDim.x + threadIdx.x;
  if (i < K * N) {
    int n = i / K, k = i - n * K;
    out[i] = (_Float16)in[(size_t)k * N + n];
  }
}

// C[M,N] = A[M,K] @ B[K,N] + bias[N] ; A row-major f16, BT = B^T row-major [N,K] f16.
// No LDS: all fragments are contiguous per-lane vector loads from global (L2/WGP$ resident).
__global__ __launch_bounds__(256) void gemm_f16f32(
    const _Float16* __restrict__ A, const _Float16* __restrict__ BT,
    const float* __restrict__ bias, float* __restrict__ C,
    int M, int N, int K) {
  const int tile_m = blockIdx.y * 64;
  const int tile_n = blockIdx.x * 128;
  const int wave = threadIdx.x >> 5;
  const int wm = (wave >> 2) * 32;   // 0 or 32
  const int wn = (wave & 3) * 32;    // 0,32,64,96
  const _Float16* Ab = A  + (size_t)(tile_m + wm) * K;
  const _Float16* Bb = BT + (size_t)(tile_n + wn) * K;
  v8f acc[2][2] = {};
  for (int k0 = 0; k0 < K; k0 += 32) {
    v16h a0 = load_a_frag (Ab + k0, K);
    v16h a1 = load_a_frag (Ab + (size_t)16 * K + k0, K);
    v16h b0 = load_bT_frag(Bb + k0, K);
    v16h b1 = load_bT_frag(Bb + (size_t)16 * K + k0, K);
    acc[0][0] = WMMA_F32_F16(a0, b0, acc[0][0]);
    acc[0][1] = WMMA_F32_F16(a0, b1, acc[0][1]);
    acc[1][0] = WMMA_F32_F16(a1, b0, acc[1][0]);
    acc[1][1] = WMMA_F32_F16(a1, b1, acc[1][1]);
  }
  const int lane = lane_id();
  const int ro = (lane & 16) ? 8 : 0;
  const int co = lane & 15;
#pragma unroll
  for (int im = 0; im < 2; ++im)
#pragma unroll
    for (int in = 0; in < 2; ++in)
#pragma unroll
      for (int j = 0; j < 8; ++j) {
        int r = tile_m + wm + im * 16 + ro + j;
        int c = tile_n + wn + in * 16 + co;
        C[(size_t)r * N + c] = acc[im][in][j] + bias[c];
      }
}

// Per-(b,h,t): layernorm q,k (head_dim 64); q,k f16 [B,H,T,64]; v f16 TRANSPOSED [B,H,64,T].
__global__ __launch_bounds__(256) void ln_split(
    const float* __restrict__ qkv,
    const float* __restrict__ gq, const float* __restrict__ bq,
    const float* __restrict__ gk, const float* __restrict__ bk,
    _Float16* __restrict__ qh, _Float16* __restrict__ kh, _Float16* __restrict__ vt) {
  int wg = blockIdx.x * 8 + (threadIdx.x >> 5);   // one wave per (b,h,t)
  const int lane = threadIdx.x & 31;
  int t = wg & 2047; wg >>= 11;
  int h = wg & 15;   wg >>= 4;
  int b = wg;
  const float* src = qkv + ((size_t)(b * 2048 + t)) * 3072;
  const size_t dst   = (((size_t)(b * 16 + h)) * 2048 + t) * 64;
  const size_t vbase = ((size_t)(b * 16 + h) * 64) * 2048 + t;
  vt[vbase + (size_t)lane * 2048]        = (_Float16)src[2048 + h * 64 + lane];
  vt[vbase + (size_t)(lane + 32) * 2048] = (_Float16)src[2048 + h * 64 + lane + 32];
#pragma unroll
  for (int w = 0; w < 2; ++w) {     // w=0: q, w=1: k
    const float* s = src + (w ? 1024 : 0) + h * 64;
    float x0 = s[lane], x1 = s[lane + 32];
    float sum = x0 + x1;
#pragma unroll
    for (int msk = 1; msk <= 16; msk <<= 1) sum += __shfl_xor(sum, msk, 32);
    float mu = sum * (1.0f / 64.0f);
    float d0 = x0 - mu, d1 = x1 - mu;
    float vs = d0 * d0 + d1 * d1;
#pragma unroll
    for (int msk = 1; msk <= 16; msk <<= 1) vs += __shfl_xor(vs, msk, 32);
    float rstd = rsqrtf(vs * (1.0f / 64.0f) + 1e-5f);
    const float* g  = w ? gk : gq;
    const float* be = w ? bk : bq;
    _Float16* dp = w ? kh : qh;
    dp[dst + lane]      = (_Float16)(d0 * rstd * g[lane]      + be[lane]);
    dp[dst + lane + 32] = (_Float16)(d1 * rstd * g[lane + 32] + be[lane + 32]);
  }
}

// Flash attention, transposed-score formulation. grid (T/128, B*H), 8 waves/block,
// each wave owns 16 query rows; each lane owns ONE query (q = lane&15).
// S^T = K_tile @ Q^T  -> C-layout: VGPR j = key, lane = query.
// O^T = V^T @ P^T     -> C-layout: VGPR j = d,   lane = query.  No LDS anywhere.
__global__ __launch_bounds__(256) void flash_attn(
    const _Float16* __restrict__ qh, const _Float16* __restrict__ kh,
    const _Float16* __restrict__ vt, _Float16* __restrict__ oh) {
  const int bh   = blockIdx.y;          // b*16 + h
  const int wave = threadIdx.x >> 5;
  const int lane = threadIdx.x & 31;
  const int q0   = blockIdx.x * 128 + wave * 16;
  const _Float16* Q  = qh + ((size_t)bh * 2048 + q0) * 64;
  const _Float16* Kp = kh + (size_t)bh * 2048 * 64;
  const _Float16* Vt = vt + (size_t)bh * 64 * 2048;
  // Q^T as B operand: col=lane=query, K halfs = d (contiguous per lane)
  v16h bq0 = load_bT_frag(Q, 64);       // d 0..31
  v16h bq1 = load_bT_frag(Q + 32, 64);  // d 32..63
  float m = -1e30f, l = 0.0f;
  v8f o[4] = {};
  const float scale = 0.125f;           // 64^-0.5
  const bool hi16 = (lane & 16) != 0;
  union Pk { v8h h; int i[4]; };
  for (int kb = 0; kb < 2048; kb += 32) {
    v8f z = {};
    // S^T tiles: keys [kb,kb+16) and [kb+16,kb+32); contraction over d=64
    v8f s0 = WMMA_F32_F16(load_a_frag(Kp + (size_t)kb * 64 + 32, 64), bq1,
             WMMA_F32_F16(load_a_frag(Kp + (size_t)kb * 64 +  0, 64), bq0, z));
    v8f s1 = WMMA_F32_F16(load_a_frag(Kp + (size_t)(kb + 16) * 64 + 32, 64), bq1,
             WMMA_F32_F16(load_a_frag(Kp + (size_t)(kb + 16) * 64 +  0, 64), bq0, z));
    // online softmax: this lane sees 16 of the 32 keys for its query;
    // the xor-16 partner lane holds the other 16.
    float c0[8], c1[8];
    float mx = m;
#pragma unroll
    for (int j = 0; j < 8; ++j) {
      c0[j] = s0[j] * scale; c1[j] = s1[j] * scale;
      mx = fmaxf(mx, fmaxf(c0[j], c1[j]));
    }
    mx = fmaxf(mx, __shfl_xor(mx, 16, 32));
    float alpha = __expf(m - mx);
    m = mx;
    Pk p0, p1;
    float sum = 0.0f;
#pragma unroll
    for (int j = 0; j < 8; ++j) {
      float e0 = __expf(c0[j] - mx), e1 = __expf(c1[j] - mx);
      sum += e0 + e1;
      p0.h[j] = (_Float16)e0; p1.h[j] = (_Float16)e1;
    }
    sum += __shfl_xor(sum, 16, 32);
    l = l * alpha + sum;
    // Build P^T B-fragment (col=query, halfs = 32 keys) with one xor-16 exchange:
    // lane<16 : halfs 0..7 = own p0 (keys kb+0..7),  8..15 = partner p0 (keys kb+8..15)
    // lane>=16: halfs 0..7 = partner p1 (keys kb+16..23), 8..15 = own p1 (keys kb+24..31)
    Pk o0, o1;
#pragma unroll
    for (int i = 0; i < 4; ++i) {
      o0.i[i] = __shfl_xor(p0.i[i], 16, 32);
      o1.i[i] = __shfl_xor(p1.i[i], 16, 32);
    }
    v16h bp;
#pragma unroll
    for (int i = 0; i < 8; ++i) {
      bp[i]     = hi16 ? o1.h[i] : p0.h[i];
      bp[8 + i] = hi16 ? p1.h[i] : o0.h[i];
    }
    // rescale running O^T (alpha is per-lane = per-query)
#pragma unroll
    for (int dt = 0; dt < 4; ++dt)
#pragma unroll
      for (int j = 0; j < 8; ++j) o[dt][j] *= alpha;
    // O^T += V^T_tile @ P^T  (V^T rows = d, contiguous keys per lane)
#pragma unroll
    for (int dt = 0; dt < 4; ++dt)
      o[dt] = WMMA_F32_F16(load_a_frag(Vt + (size_t)(dt * 16) * 2048 + kb, 2048), bp, o[dt]);
  }
  // epilogue: lane's query t = q0 + (lane&15); d = dt*16 + ro + j  -> packed 16B stores
  const int b = bh >> 4, h = bh & 15;
  const int co = lane & 15;
  const int ro = hi16 ? 8 : 0;
  const float inv = 1.0f / l;
  _Float16* orow = oh + ((size_t)(b * 2048 + q0 + co)) * 1024 + h * 64 + ro;
#pragma unroll
  for (int dt = 0; dt < 4; ++dt) {
    v8h pk;
#pragma unroll
    for (int j = 0; j < 8; ++j) pk[j] = (_Float16)(o[dt][j] * inv);
    *(v8h*)(orow + dt * 16) = pk;
  }
}

extern "C" void kernel_launch(void* const* d_in, const int* in_sizes, int n_in,
                              void* d_out, int out_size, void* d_ws, size_t ws_size,
                              hipStream_t stream) {
  const float* x     = (const float*)d_in[0];
  const float* w_qkv = (const float*)d_in[1];
  const float* b_qkv = (const float*)d_in[2];
  const float* g_q   = (const float*)d_in[3];
  const float* be_q  = (const float*)d_in[4];
  const float* g_k   = (const float*)d_in[5];
  const float* be_k  = (const float*)d_in[6];
  const float* w_out = (const float*)d_in[7];
  const float* b_out = (const float*)d_in[8];

  char* ws = (char*)d_ws;
  const size_t MB = 1ull << 20;
  _Float16* x_h     = (_Float16*)(ws + 0);        // 8 MB : [4096,1024]
  _Float16* wqkvT_h = (_Float16*)(ws + 8  * MB);  // 6 MB : [3072,1024] (= w_qkv^T)
  _Float16* woutT_h = (_Float16*)(ws + 14 * MB);  // 2 MB : [1024,1024] (= w_out^T)
  float*    qkv     = (float*)   (ws + 16 * MB);  // 48 MB: [4096,3072] f32
  _Float16* q_h     = (_Float16*)(ws + 64 * MB);  // 8 MB : [2,16,2048,64]
  _Float16* k_h     = (_Float16*)(ws + 72 * MB);  // 8 MB : [2,16,2048,64]
  _Float16* v_t     = (_Float16*)(ws + 80 * MB);  // 8 MB : [2,16,64,2048] (transposed)
  _Float16* at_h    = (_Float16*)(ws + 88 * MB);  // 8 MB : [4096,1024]
  (void)in_sizes; (void)n_in; (void)out_size; (void)ws_size;

  cvt_f32_to_f16<<<4194304 / 256, 256, 0, stream>>>(x, x_h, 4194304);
  cvt_transpose_f16<<<3145728 / 256, 256, 0, stream>>>(w_qkv, wqkvT_h, 1024, 3072);
  cvt_transpose_f16<<<1048576 / 256, 256, 0, stream>>>(w_out, woutT_h, 1024, 1024);

  // qkv = x @ w_qkv + b_qkv   (M=4096, N=3072, K=1024)
  gemm_f16f32<<<dim3(3072 / 128, 4096 / 64), 256, 0, stream>>>(
      x_h, wqkvT_h, b_qkv, qkv, 4096, 3072, 1024);

  // per-head layernorm on q,k + pack q/k f16 [B,H,T,64], v f16 [B,H,64,T]
  ln_split<<<65536 / 8, 256, 0, stream>>>(qkv, g_q, be_q, g_k, be_k, q_h, k_h, v_t);

  // flash attention -> at_h [B,T,C] f16
  flash_attn<<<dim3(2048 / 128, 32), 256, 0, stream>>>(q_h, k_h, v_t, at_h);

  // out = at @ w_out + b_out  (M=4096, N=1024, K=1024) -> d_out f32
  gemm_f16f32<<<dim3(1024 / 128, 4096 / 64), 256, 0, stream>>>(
      at_h, woutT_h, b_out, (float*)d_out, 4096, 1024, 1024);
}
